// Attention_86792699117843
// MI455X (gfx1250) — compile-verified, matrix operand order checked by reading, not run on hardware
//
#include <hip/hip_runtime.h>
#include <hip/hip_bf16.h>

typedef __bf16 bf16;
typedef __attribute__((ext_vector_type(16))) __bf16 v16bf;
typedef __attribute__((ext_vector_type(8)))  float  v8f;

static constexpr int B_ = 2, S_ = 2048, D_ = 1024, H_ = 16, KV_ = 4, HD_ = 64;

// ---------- bf16 helpers (RNE, bit-level) ----------
__device__ inline bf16 tobf(float f) {
  unsigned u = __builtin_bit_cast(unsigned, f);
  unsigned r = u + 0x7fffu + ((u >> 16) & 1u);
  unsigned short h = (unsigned short)(r >> 16);
  return __builtin_bit_cast(bf16, h);
}
__device__ inline float frombf(bf16 h) {
  unsigned short s = __builtin_bit_cast(unsigned short, h);
  unsigned u = ((unsigned)s) << 16;
  return __builtin_bit_cast(float, u);
}

// ---------- WMMA ----------
__device__ inline v8f wmma_bf16(v16bf a, v16bf b, v8f c) {
  return __builtin_amdgcn_wmma_f32_16x16x32_bf16(false, a, false, b, (short)0, c, false, false);
}

// A/B fragment from a row-major matrix: caller passes (row base + k0).
__device__ inline v16bf frag_k(const bf16* rowp) {
  int lane = threadIdx.x & 31;
  int hi = (lane & 16) ? 8 : 0;
  union { uint4 u[2]; v16bf v; } c;
  c.u[0] = *reinterpret_cast<const uint4*>(rowp + hi);
  c.u[1] = *reinterpret_cast<const uint4*>(rowp + 16 + hi);
  return c.v;
}
// B fragment from pre-swizzled weights: 512 bf16 per fragment, lane-contiguous.
__device__ inline v16bf frag_swz(const bf16* fb) {
  int lane = threadIdx.x & 31;
  const uint4* p = reinterpret_cast<const uint4*>(fb + lane * 16);
  union { uint4 u[2]; v16bf v; } c;
  c.u[0] = p[0]; c.u[1] = p[1];
  return c.v;
}

// ---------- conversions / swizzles ----------
__global__ void k_cvt_x(const float* __restrict__ x, bf16* __restrict__ xb, int n) {
  int i = blockIdx.x * blockDim.x + threadIdx.x;
  if (i < n) xb[i] = tobf(x[i]);
}

__device__ inline int kmap(int lane, int e) {
  int hi = (lane & 16) ? 8 : 0;
  return (e < 8) ? (hi + e) : (16 + hi + (e - 8));
}

// Concatenated Wq|Wk|Wv (K=1024, N=1536) -> B-fragment order (96 ntiles x 32 ktiles x 512)
__global__ void k_swz_qkv(const float* __restrict__ Wq, const float* __restrict__ Wk,
                          const float* __restrict__ Wv, bf16* __restrict__ wqkv) {
  int idx = blockIdx.x * blockDim.x + threadIdx.x;
  if (idx >= 96 * 32 * 512) return;
  int e = idx & 15, lane = (idx >> 4) & 31, fk = idx >> 9;
  int kt = fk & 31, nt = fk >> 5;
  int k = kt * 32 + kmap(lane, e);
  int c = nt * 16 + (lane & 15);
  float v;
  if (c < 1024)      v = Wq[k * 1024 + c];
  else if (c < 1280) v = Wk[k * 256 + (c - 1024)];
  else               v = Wv[k * 256 + (c - 1280)];
  wqkv[idx] = tobf(v);
}

// Wo (K=1024, N=1024) -> B-fragment order (64 ntiles x 32 ktiles x 512)
__global__ void k_swz_wo(const float* __restrict__ Wo, bf16* __restrict__ wob) {
  int idx = blockIdx.x * blockDim.x + threadIdx.x;
  if (idx >= 64 * 32 * 512) return;
  int e = idx & 15, lane = (idx >> 4) & 31, fk = idx >> 9;
  int kt = fk & 31, nt = fk >> 5;
  int k = kt * 32 + kmap(lane, e);
  int n = nt * 16 + (lane & 15);
  wob[idx] = tobf(Wo[k * 1024 + n]);
}

// ---------- QKV projection GEMM ----------
// Block = 8 waves arranged 4(M) x 2(N): 128 rows x 192 cols of output.
// Wave  = 2(M) x 6(N) register tile: 32 rows x 96 cols -> 12 accumulators.
// Each B fragment feeds 2 WMMAs; weight panel read once per 128-row block.
__global__ void __launch_bounds__(256) k_qkv(const bf16* __restrict__ xb,
                                             const bf16* __restrict__ wqkv,
                                             bf16* __restrict__ qb, bf16* __restrict__ kb,
                                             bf16* __restrict__ vt) {
  int lane = threadIdx.x & 31, wv = threadIdx.x >> 5;
  int mb = blockIdx.x & 31, nb = blockIdx.x >> 5;   // 32 x 8 blocks
  int wm = wv & 3, wn = wv >> 2;
  int mbase  = mb * 128 + wm * 32;
  int ntbase = nb * 12 + wn * 6;
  int jl = lane & 15, hi8 = (lane & 16) ? 8 : 0;
  const bf16* xrow0 = xb + (mbase + jl) * 1024;
  const bf16* xrow1 = xb + (mbase + 16 + jl) * 1024;

  v8f acc[2][6] = {};
  for (int kt = 0; kt < 32; ++kt) {
    v16bf a0 = frag_k(xrow0 + kt * 32);
    v16bf a1 = frag_k(xrow1 + kt * 32);
#pragma unroll
    for (int j = 0; j < 6; ++j) {
      v16bf bw = frag_swz(wqkv + ((ntbase + j) * 32 + kt) * 512);
      acc[0][j] = wmma_bf16(a0, bw, acc[0][j]);
      acc[1][j] = wmma_bf16(a1, bw, acc[1][j]);
    }
  }
#pragma unroll
  for (int t = 0; t < 2; ++t)
#pragma unroll
    for (int j = 0; j < 6; ++j) {
      int c = (ntbase + j) * 16 + jl;
#pragma unroll
      for (int r = 0; r < 8; ++r) {
        int rbs = mbase + t * 16 + r + hi8;
        int b = rbs >> 11, s = rbs & (S_ - 1);
        bf16 v = tobf(acc[t][j][r]);
        if (c < 1024) {
          int h = c >> 6, d = c & 63;
          qb[(((b * H_ + h) * S_) + s) * HD_ + d] = v;
        } else if (c < 1280) {
          int cc = c - 1024; int g = cc >> 6, d = cc & 63;
          kb[(((b * KV_ + g) * S_) + s) * HD_ + d] = v;
        } else {
          int cc = c - 1280; int g = cc >> 6, d = cc & 63;
          vt[(((b * KV_ + g) * HD_) + d) * S_ + s] = v;   // V stored transposed
        }
      }
    }
}

// ---------- RoPE in place on Q and K (pairs d, d+32; cos/sin halves identical) ----------
__global__ void k_rope(bf16* __restrict__ qb, bf16* __restrict__ kb,
                       const float* __restrict__ cosT, const float* __restrict__ sinT) {
  int idx = blockIdx.x * blockDim.x + threadIdx.x;
  const int nq = B_ * H_ * S_ * 32;
  const int nk = B_ * KV_ * S_ * 32;
  if (idx >= nq + nk) return;
  bf16* buf; int base, s, d;
  if (idx < nq) {
    d = idx & 31; s = (idx >> 5) & (S_ - 1);
    int rest = idx >> 16; int h = rest & 15, b = rest >> 4;
    buf = qb; base = (((b * H_ + h) * S_) + s) * HD_;
  } else {
    int j = idx - nq;
    d = j & 31; s = (j >> 5) & (S_ - 1);
    int rest = j >> 16; int g = rest & 3, b = rest >> 2;
    buf = kb; base = (((b * KV_ + g) * S_) + s) * HD_;
  }
  float c = cosT[s * HD_ + d], sn = sinT[s * HD_ + d];
  float lo = frombf(buf[base + d]), hiv = frombf(buf[base + d + 32]);
  buf[base + d]      = tobf(lo * c - hiv * sn);
  buf[base + d + 32] = tobf(hiv * c + lo * sn);
}

// ---------- flash attention: one wave per 16-query tile, 32 keys / iteration ----------
__global__ void __launch_bounds__(256) k_attn(const bf16* __restrict__ qb,
                                              const bf16* __restrict__ kb,
                                              const bf16* __restrict__ vt,
                                              bf16* __restrict__ ob) {
  __shared__ __align__(16) bf16 plds[8 * 16 * 32];   // 1 KB per wave
  int lane = threadIdx.x & 31, wv = threadIdx.x >> 5;
  int wg = blockIdx.x * 8 + wv;
  int qt = wg & (S_ / 16 - 1);
  int h  = (wg >> 7) & (H_ - 1);
  int b  = wg >> 11;
  int g  = h / (H_ / KV_);
  int q0 = qt * 16;
  int jl = lane & 15, hi8 = (lane & 16) ? 8 : 0;
  const bf16* qbh = qb + ((b * H_ + h) * S_) * HD_;
  const bf16* kbg = kb + ((b * KV_ + g) * S_) * HD_;
  const bf16* vtg = vt + ((b * KV_ + g) * HD_) * S_;
  bf16* myLds = plds + wv * 512;

  v16bf aq[2];
  aq[0] = frag_k(qbh + (q0 + jl) * HD_ + 0);
  aq[1] = frag_k(qbh + (q0 + jl) * HD_ + 32);

  float mrow[8], lrow[8];
#pragma unroll
  for (int r = 0; r < 8; ++r) { mrow[r] = -3.0e38f; lrow[r] = 0.0f; }
  v8f o[4] = {};

  int ktmax = (q0 + 15) >> 5;
  for (int kt = 0; kt <= ktmax; ++kt) {
    int j0 = kt * 32;
    v8f s0 = {}, s1 = {};
#pragma unroll
    for (int dk = 0; dk < 2; ++dk) {
      v16bf bk0 = frag_k(kbg + (j0 + jl) * HD_ + dk * 32);
      s0 = wmma_bf16(aq[dk], bk0, s0);
      v16bf bk1 = frag_k(kbg + (j0 + 16 + jl) * HD_ + dk * 32);
      s1 = wmma_bf16(aq[dk], bk1, s1);
    }
    __builtin_prefetch(kbg + (j0 + 32 + jl) * HD_, 0, 0);

    float p0v[8], p1v[8], scl[8];
#pragma unroll
    for (int r = 0; r < 8; ++r) {
      int qrow = q0 + r + hi8;
      float x0 = s0[r] * 0.125f; x0 = (j0 + jl > qrow) ? -3.0e38f : x0;
      float x1 = s1[r] * 0.125f; x1 = (j0 + 16 + jl > qrow) ? -3.0e38f : x1;
      float mx = fmaxf(x0, x1);
#pragma unroll
      for (int msk = 1; msk < 16; msk <<= 1) mx = fmaxf(mx, __shfl_xor(mx, msk, 32));
      float mnew = fmaxf(mrow[r], mx);
      float sc = __expf(mrow[r] - mnew);
      float p0 = __expf(x0 - mnew);
      float p1 = __expf(x1 - mnew);
      float rs = p0 + p1;
#pragma unroll
      for (int msk = 1; msk < 16; msk <<= 1) rs += __shfl_xor(rs, msk, 32);
      lrow[r] = lrow[r] * sc + rs;
      mrow[r] = mnew;
      scl[r] = sc;
      p0v[r] = p0; p1v[r] = p1;
    }
#pragma unroll
    for (int f = 0; f < 4; ++f)
#pragma unroll
      for (int r = 0; r < 8; ++r) o[f][r] *= scl[r];

    // transpose P (C-layout) -> A-layout via LDS
#pragma unroll
    for (int r = 0; r < 8; ++r) {
      myLds[(r + hi8) * 32 + jl]      = tobf(p0v[r]);
      myLds[(r + hi8) * 32 + 16 + jl] = tobf(p1v[r]);
    }
    v16bf pa = frag_k(myLds + jl * 32);
#pragma unroll
    for (int f = 0; f < 4; ++f) {
      v16bf bv = frag_k(vtg + (f * 16 + jl) * S_ + j0);   // contiguous over keys
      o[f] = wmma_bf16(pa, bv, o[f]);
    }
  }
#pragma unroll
  for (int f = 0; f < 4; ++f)
#pragma unroll
    for (int r = 0; r < 8; ++r) {
      int srow = q0 + r + hi8;
      ob[(b * S_ + srow) * 1024 + h * HD_ + f * 16 + jl] = tobf(o[f][r] / lrow[r]);
    }
}

// ---------- output projection GEMM ----------
// Block = 8 waves arranged 4(M) x 2(N): 128 x 128 output. Wave = 2(M) x 4(N).
__global__ void __launch_bounds__(256) k_oproj(const bf16* __restrict__ ob,
                                               const bf16* __restrict__ wob,
                                               float* __restrict__ out) {
  int lane = threadIdx.x & 31, wv = threadIdx.x >> 5;
  int mb = blockIdx.x & 31, nb = blockIdx.x >> 5;   // 32 x 8 blocks
  int wm = wv & 3, wn = wv >> 2;
  int mbase  = mb * 128 + wm * 32;
  int ntbase = nb * 8 + wn * 4;
  int jl = lane & 15, hi8 = (lane & 16) ? 8 : 0;
  const bf16* xrow0 = ob + (mbase + jl) * 1024;
  const bf16* xrow1 = ob + (mbase + 16 + jl) * 1024;

  v8f acc[2][4] = {};
  for (int kt = 0; kt < 32; ++kt) {
    v16bf a0 = frag_k(xrow0 + kt * 32);
    v16bf a1 = frag_k(xrow1 + kt * 32);
#pragma unroll
    for (int j = 0; j < 4; ++j) {
      v16bf bw = frag_swz(wob + ((ntbase + j) * 32 + kt) * 512);
      acc[0][j] = wmma_bf16(a0, bw, acc[0][j]);
      acc[1][j] = wmma_bf16(a1, bw, acc[1][j]);
    }
  }
#pragma unroll
  for (int t = 0; t < 2; ++t)
#pragma unroll
    for (int j = 0; j < 4; ++j)
#pragma unroll
      for (int r = 0; r < 8; ++r)
        out[(mbase + t * 16 + r + hi8) * 1024 + (ntbase + j) * 16 + jl] = acc[t][j][r];
}

extern "C" void kernel_launch(void* const* d_in, const int* in_sizes, int n_in,
                              void* d_out, int out_size, void* d_ws, size_t ws_size,
                              hipStream_t stream) {
  (void)in_sizes; (void)n_in; (void)out_size; (void)ws_size;
  const float* x    = (const float*)d_in[0];
  const float* cosT = (const float*)d_in[1];
  const float* sinT = (const float*)d_in[2];
  const float* Wq   = (const float*)d_in[3];
  const float* Wk   = (const float*)d_in[4];
  const float* Wv   = (const float*)d_in[5];
  const float* Wo   = (const float*)d_in[6];
  float* out = (float*)d_out;

  char* ws = (char*)d_ws;
  bf16* xb   = (bf16*)(ws);               // 8 MB
  bf16* wqkv = (bf16*)(ws + 8388608);     // 3 MB
  bf16* wob  = (bf16*)(ws + 11534336);    // 2 MB
  bf16* qb   = (bf16*)(ws + 13631488);    // 8 MB  [B][H][S][64]
  bf16* kb   = (bf16*)(ws + 22020096);    // 2 MB  [B][KV][S][64]
  bf16* vt   = (bf16*)(ws + 24117248);    // 2 MB  [B][KV][64][S]
  bf16* ob   = (bf16*)(ws + 26214400);    // 8 MB  [B*S][1024]

  k_cvt_x  <<<16384, 256, 0, stream>>>(x, xb, 4096 * 1024);
  k_swz_qkv<<< 6144, 256, 0, stream>>>(Wq, Wk, Wv, wqkv);
  k_swz_wo <<< 4096, 256, 0, stream>>>(Wo, wob);
  k_qkv    <<<  256, 256, 0, stream>>>(xb, wqkv, qb, kb, vt);
  k_rope   <<<10240, 256, 0, stream>>>(qb, kb, cosT, sinT);
  k_attn   <<<  512, 256, 0, stream>>>(qb, kb, vt, ob);
  k_oproj  <<<  256, 256, 0, stream>>>(ob, wob, out);
}
